// ConvNet_baseNonlinearHead_new_81192061763838
// MI455X (gfx1250) — matrix operand order repaired and checked by reading, not compiled, not run on hardware
//
#include <hip/hip_runtime.h>
#include <math.h>

// ---------------- problem constants ----------------
#define B_TOT   512
#define NCH     62          // input channels
#define T_LEN   2000        // time length
#define NSF     16          // spatial filters (NS)
#define NTF     16          // time filters (NT)
#define L_FILT  25          // time filter taps
#define PADT    12          // (L-1)/2
#define POOL    125
#define WPOOL   16          // 2000/125
#define CHUNK   512         // T streamed in chunks through LDS
#define SBUF_STRIDE 576     // CHUNK + 48 halo, padded

typedef _Float16 v16h __attribute__((ext_vector_type(16)));
typedef float    v8f  __attribute__((ext_vector_type(8)));

// A-matrix (16x32 f16) K index for element e of the v16h fragment, per ISA 7.12.2:
// lanes 0-15: VGPR0..3 hold K=0..7, VGPR4..7 hold K=16..23; lanes 16-31: +8.
__device__ __forceinline__ int a_k_of(int e, int lane) {
    return ((e < 8) ? 0 : 16) + (((e >> 1) & 3) << 1) + (e & 1) + ((lane >> 4) << 3);
}

// CDNA5 async global->LDS copy (VFLAT GLOBAL_LOAD_ASYNC_TO_LDS_B128, tracked by ASYNCcnt).
// VDST VGPR carries the LDS byte offset; VADDR pair carries the 64-bit global address.
__device__ __forceinline__ void async_copy_b128(void* lds_dst, const void* gsrc) {
    unsigned loff = (unsigned)(unsigned long long)
        ((__attribute__((address_space(3))) char*)lds_dst);
    asm volatile("global_load_async_to_lds_b128 %0, %1, off"
                 :: "v"(loff), "v"((unsigned long long)gsrc)
                 : "memory");
}

__device__ __forceinline__ void wait_async0() {
#if __has_builtin(__builtin_amdgcn_s_wait_asynccnt)
    __builtin_amdgcn_s_wait_asynccnt(0);
#else
    asm volatile("s_wait_asynccnt 0x0" ::: "memory");
#endif
}

// =====================================================================
// Kernel 1: fused spatial GEMM -> time conv GEMM -> ELU -> avgpool(125)
// One workgroup (4 waves) per batch sample; T streamed in LDS chunks.
// x tiles are staged via async global->LDS b128 copies (per-wave slot,
// synchronized with the wave-private ASYNCcnt -> no workgroup barrier).
// =====================================================================
__global__ __launch_bounds__(128)
void fused_conv_pool_kernel(const float* __restrict__ x,
                            const float* __restrict__ Ws,
                            const float* __restrict__ bs,
                            const float* __restrict__ Wt,
                            const float* __restrict__ bt,
                            float* __restrict__ p_raw,
                            float* __restrict__ gsum,
                            float* __restrict__ gsumsq)
{
    __shared__ _Float16 s_buf[NSF][SBUF_STRIDE];            // f16 spatial-conv chunk (+halo)
    __shared__ float    pool_acc[NTF][NSF][WPOOL];          // pooled ELU accumulators
    __shared__ __align__(16) float x_stage[4][64][16];      // per-wave async x tile (62 rows used)

    const int tid  = threadIdx.x;
    const int lane = tid & 31;
    const int wave = tid >> 5;
    const int b    = blockIdx.x;
    const int mn   = lane & 15;          // M for A rows, N for B/C cols
    const int hi   = lane >> 4;          // lane-half select

    // ---- zero pool accumulators ----
    for (int i = tid; i < NTF * NSF * WPOOL; i += 128)
        ((float*)pool_acc)[i] = 0.0f;

    // ---- stage-1 A fragments: Ws (16 x 62, K padded to 64 -> two 16x32 frags) ----
    v16h a1_0, a1_1;
    #pragma unroll
    for (int e = 0; e < 16; ++e) {
        int k0 = a_k_of(e, lane);
        int k1 = k0 + 32;
        a1_0[e] = (_Float16)Ws[mn * NCH + k0];                       // k0 in [0,31] < 62 always
        a1_1[e] = (_Float16)((k1 < NCH) ? Ws[mn * NCH + k1] : 0.0f); // zero-pad K 62..63
    }
    // ---- stage-2 A fragment: Wt (16 x 25, K padded to 32) ----
    v16h a2;
    #pragma unroll
    for (int e = 0; e < 16; ++e) {
        int k = a_k_of(e, lane);
        a2[e] = (_Float16)((k < L_FILT) ? Wt[mn * L_FILT + k] : 0.0f);
    }
    // per-lane output-row biases (rows r+8*hi of the C tile)
    float bs_r[8], bt_r[8];
    #pragma unroll
    for (int r = 0; r < 8; ++r) {
        bs_r[r] = bs[r + (hi << 3)];
        bt_r[r] = bt[r + (hi << 3)];
    }

    const float* xb = x + (size_t)b * NCH * T_LEN;

    for (int cs = 0; cs < T_LEN; cs += CHUNK) {
        const int clen = (T_LEN - cs < CHUNK) ? (T_LEN - cs) : CHUNK;
        const int base = cs - 16;                 // buffer col = tau - base

        __syncthreads();                          // previous chunk fully consumed
        // zero s_buf (covers conv zero-padding halos and K-pad slack)
        for (int i = tid; i < NSF * SBUF_STRIDE; i += 128)
            ((_Float16*)s_buf)[i] = (_Float16)0.0f;
        __syncthreads();

        // ---------- stage 1: s = Ws @ x  (+bs), write f16 into LDS ----------
        const int tlo = (cs >> 4) > 0 ? (cs >> 4) - 1 : 0;
        int thi = (cs + clen + 34) >> 4; if (thi > 125) thi = 125;
        for (int tt = tlo + wave; tt < thi; tt += 4) {
            const int t0  = tt * 16;
            __builtin_prefetch(xb + t0 + 64, 0, 1);    // global_prefetch_b8: next tile of this wave

            // async-stage x[0..61][t0..t0+15] into this wave's LDS slot:
            // 62 rows x 4 b128 segments = 248 lane-transfers -> 8 async issues.
            #pragma unroll
            for (int base_id = 0; base_id < NCH * 4; base_id += 32) {
                int id = base_id + lane;
                if (id < NCH * 4) {
                    int k = id >> 2, seg = id & 3;
                    async_copy_b128(&x_stage[wave][k][seg * 4],
                                    xb + (size_t)k * T_LEN + t0 + seg * 4);
                }
            }
            wait_async0();                              // wave-private ASYNCcnt

            // build B fragments from the staged tile (f32 -> f16)
            v16h b0, b1;
            #pragma unroll
            for (int e = 0; e < 16; ++e) {
                int k0 = ((e < 8) ? 0 : 16) + (e & 7) + (hi << 3);   // B layout matches A: lanes 0-15 K=0..7,16..23; 16-31 +8
                int k1 = k0 + 32;
                b0[e] = (_Float16)x_stage[wave][k0][mn];
                b1[e] = (_Float16)((k1 < NCH) ? x_stage[wave][k1][mn] : 0.0f);
            }
            v8f c = {};
            c = __builtin_amdgcn_wmma_f32_16x16x32_f16(false, a1_0, false, b0, (short)0, c, false, false);
            c = __builtin_amdgcn_wmma_f32_16x16x32_f16(false, a1_1, false, b1, (short)0, c, false, false);
            const int scol = t0 + mn - base;
            #pragma unroll
            for (int r = 0; r < 8; ++r)
                s_buf[r + (hi << 3)][scol] = (_Float16)(c[r] + bs_r[r]);
        }
        __syncthreads();

        // ---------- stage 2: t = Wt (x) s, ELU, pool ----------
        const int ntiles = clen >> 4;             // clen is a multiple of 16
        for (int u = wave; u < NSF * ntiles; u += 4) {
            const int ns = u / ntiles;
            const int t0 = cs + (u - ns * ntiles) * 16;
            v16h bf;
            #pragma unroll
            for (int e = 0; e < 16; ++e) {
                int k = ((e < 8) ? 0 : 16) + (e & 7) + (hi << 3);   // K of the shifted-window B matrix
                bf[e] = s_buf[ns][t0 + mn + k - PADT - base];
            }
            v8f c = {};
            c = __builtin_amdgcn_wmma_f32_16x16x32_f16(false, a2, false, bf, (short)0, c, false, false);
            const int wp = (t0 + mn) / POOL;
            #pragma unroll
            for (int r = 0; r < 8; ++r) {
                float v = c[r] + bt_r[r];
                v = (v > 0.0f) ? v : (__expf(v) - 1.0f);   // ELU(alpha=1)
                atomicAdd(&pool_acc[r + (hi << 3)][ns][wp], v);
            }
        }
    }
    __syncthreads();

    // ---------- write pooled tensor + accumulate group stats ----------
    const float inv_pool = 1.0f / (float)POOL;
    float* pb = p_raw + (size_t)b * (NTF * NSF * WPOOL);
    for (int i = tid; i < NTF * NSF * WPOOL; i += 128)
        pb[i] = ((const float*)pool_acc)[i] * inv_pool;

    const int g = b >> 8;                          // 2 groups of 256 samples
    for (int f = tid; f < NTF * NSF; f += 128) {
        float s1 = 0.0f, s2 = 0.0f;
        #pragma unroll
        for (int wp = 0; wp < WPOOL; ++wp) {
            float v = pool_acc[f >> 4][f & 15][wp] * inv_pool;
            s1 += v; s2 += v * v;
        }
        atomicAdd(&gsum[g * 256 + f], s1);
        atomicAdd(&gsumsq[g * 256 + f], s2);
    }
}

// =====================================================================
// Kernel 0: zero the global stats accumulators (graph-replay safe)
// =====================================================================
__global__ void zero_stats_kernel(float* stats) {
    int i = blockIdx.x * blockDim.x + threadIdx.x;
    if (i < 1024) stats[i] = 0.0f;
}

// =====================================================================
// Kernel 2: finalize stratified-layernorm stats (ddof=1)
// =====================================================================
__global__ void finalize_stats_kernel(const float* __restrict__ gsum,
                                      const float* __restrict__ gsumsq,
                                      float* __restrict__ mu,
                                      float* __restrict__ inv) {
    int i = blockIdx.x * blockDim.x + threadIdx.x;
    if (i < 512) {
        const float cnt = 256.0f * (float)WPOOL;   // samples * time per feature
        float m  = gsum[i] / cnt;
        float var = (gsumsq[i] - cnt * m * m) / (cnt - 1.0f);
        var = var > 0.0f ? var : 0.0f;
        mu[i]  = m;
        inv[i] = 1.0f / (sqrtf(var) + 0.001f);
    }
}

// =====================================================================
// Kernel 3: normalize, time-mean, top-8 over NT per NS, gather, emit
// =====================================================================
__global__ __launch_bounds__(256)
void norm_topk_gather_kernel(const float* __restrict__ p_raw,
                             const float* __restrict__ mu,
                             const float* __restrict__ inv,
                             float* __restrict__ out_sel,
                             float* __restrict__ out_idx) {
    __shared__ float p_s[NTF * NSF * WPOOL];
    __shared__ float m_s[NTF][NSF];
    __shared__ int   idx_s[8][NSF];

    const int tid = threadIdx.x;
    const int b   = blockIdx.x;
    const int g   = b >> 8;
    const float* pb = p_raw + (size_t)b * (NTF * NSF * WPOOL);

    for (int i = tid; i < NTF * NSF * WPOOL; i += 256) {
        int f = i >> 4;
        p_s[i] = (pb[i] - mu[g * 256 + f]) * inv[g * 256 + f];
    }
    __syncthreads();

    if (tid < NTF * NSF) {
        float s = 0.0f;
        #pragma unroll
        for (int wp = 0; wp < WPOOL; ++wp) s += p_s[tid * WPOOL + wp];
        m_s[tid >> 4][tid & 15] = s * (1.0f / (float)WPOOL);
    }
    __syncthreads();

    if (tid < NSF) {                                 // one thread per H (ns) column
        const int ns = tid;
        unsigned used = 0;
        for (int i = 0; i < 8; ++i) {                // stable top-8 (strict >, low index wins ties)
            float best = -3.0e38f; int bi = 0;
            for (int c = 0; c < NTF; ++c) {
                if (!((used >> c) & 1u) && m_s[c][ns] > best) { best = m_s[c][ns]; bi = c; }
            }
            used |= 1u << bi;
            idx_s[i][ns] = bi;
        }
    }
    __syncthreads();

    for (int i = tid; i < 8 * NSF; i += 256)          // indices (B, 8, 16), as float
        out_idx[(size_t)b * 128 + i] = (float)idx_s[i >> 4][i & 15];

    for (int v = tid; v < 8 * NSF * WPOOL; v += 256) { // sel (B, 8, 16, 16) flattened
        int i = v >> 8, rest = v & 255;
        out_sel[(size_t)b * 2048 + v] = p_s[idx_s[i][0] * 256 + rest];
    }
}

// =====================================================================
extern "C" void kernel_launch(void* const* d_in, const int* in_sizes, int n_in,
                              void* d_out, int out_size, void* d_ws, size_t ws_size,
                              hipStream_t stream) {
    const float* x  = (const float*)d_in[0];   // (512,1,62,2000)
    const float* Ws = (const float*)d_in[1];   // (16,62)
    const float* bs = (const float*)d_in[2];   // (16,)
    const float* Wt = (const float*)d_in[3];   // (16,1,1,25)
    const float* bt = (const float*)d_in[4];   // (16,)
    float* out = (float*)d_out;

    float* p_raw  = (float*)d_ws;                                   // 512*4096 f32 = 8 MB
    float* stats  = p_raw + (size_t)B_TOT * NTF * NSF * WPOOL;      // 4 x 512 f32
    float* gsum   = stats;
    float* gsumsq = stats + 512;
    float* mu     = stats + 1024;
    float* inv    = stats + 1536;

    zero_stats_kernel<<<4, 256, 0, stream>>>(stats);
    fused_conv_pool_kernel<<<B_TOT, 128, 0, stream>>>(x, Ws, bs, Wt, bt,
                                                      p_raw, gsum, gsumsq);
    finalize_stats_kernel<<<2, 256, 0, stream>>>(gsum, gsumsq, mu, inv);
    norm_topk_gather_kernel<<<B_TOT, 256, 0, stream>>>(p_raw, mu, inv,
                                                       out, out + (size_t)B_TOT * 2048);
}
